// FEELModel_87608742904139
// MI455X (gfx1250) — compile-verified, matrix operand order checked by reading, not run on hardware
//
#include <hip/hip_runtime.h>

typedef __attribute__((ext_vector_type(16))) __bf16 bf16x16;
typedef __attribute__((ext_vector_type(8)))  float  f32x8;

#define LSTR 40        // LDS row stride (elements): 80B rows keep 8-elem groups 16B aligned
#ifndef USE_ASYNC_LDS
#define USE_ASYNC_LDS 1
#endif

__device__ __forceinline__ float sigf(float x) { return 1.f / (1.f + __expf(-x)); }

#if USE_ASYNC_LDS
// CDNA5 async global->LDS copy (ASYNCcnt-tracked). VDST = LDS byte address,
// VADDR = 64-bit global address (GV mode).
__device__ __forceinline__ void async_b128(unsigned ldsAddr, const void* gaddr) {
    asm volatile("global_load_async_to_lds_b128 %0, %1, off"
                 :: "v"(ldsAddr), "v"((unsigned long long)(size_t)gaddr)
                 : "memory");
}
__device__ __forceinline__ void wait_async0() {
    asm volatile("s_wait_asynccnt 0" ::: "memory");
}
#endif

// ---------------------------------------------------------------------------
// fp32 [K][N] -> bf16 [N][K] transposed weight conversion (done once per call)
__global__ __launch_bounds__(256)
void k_f2bf_t(const float* __restrict__ src, __bf16* __restrict__ dst, int K, int N) {
    const int n = blockIdx.x * 16 + (threadIdx.x & 15);
    const int k = blockIdx.y * 16 + (threadIdx.x >> 4);
    dst[(size_t)n * K + k] = (__bf16)src[(size_t)k * N + n];
}

// biasIOU = bioux + biouh (leaves see both since h_sum == 0 still adds biouh)
__global__ void k_biasadd(const float* __restrict__ a, const float* __restrict__ b,
                          float* __restrict__ d, int n) {
    int i = blockIdx.x * blockDim.x + threadIdx.x;
    if (i < n) d[i] = a[i] + b[i];
}

// ---------------------------------------------------------------------------
// Attribute pools: POOL[w][b][512] = mean over 64 tokens of emb rows
__global__ __launch_bounds__(128)
void k_attr_pool(const float* __restrict__ emb,
                 const int* __restrict__ t0, const int* __restrict__ t1,
                 const int* __restrict__ t2, const int* __restrict__ t3,
                 const int* __restrict__ t4, const int* __restrict__ t5,
                 const int* __restrict__ t6, float* __restrict__ POOL) {
    const int b = blockIdx.x, w = blockIdx.y;
    const int* tabs[7] = {t0, t1, t2, t3, t4, t5, t6};
    const int* ids = tabs[w] + b * 64;
    const int c = threadIdx.x * 4;
    float a0 = 0, a1 = 0, a2 = 0, a3 = 0;
    for (int t = 0; t < 64; ++t) {
        if (t + 4 < 64)
            __builtin_prefetch(emb + (size_t)ids[t + 4] * 512 + c, 0, 0);
        const float4 e = *(const float4*)(emb + (size_t)ids[t] * 512 + c);
        a0 += e.x; a1 += e.y; a2 += e.z; a3 += e.w;
    }
    float* o = POOL + ((size_t)w * 512 + b) * 512 + c;
    const float s = 1.f / 64.f;
    o[0] = a0 * s; o[1] = a1 * s; o[2] = a2 * s; o[3] = a3 * s;
}

// Sequence pools -> bf16 node vectors X[tree][b][node][512] (mean over 32 tokens)
__global__ __launch_bounds__(128)
void k_seq_pool(const float* __restrict__ emb,
                const int* __restrict__ q, const int* __restrict__ p,
                const int* __restrict__ n, __bf16* __restrict__ X) {
    const int b = blockIdx.x, node = blockIdx.y, tree = blockIdx.z;
    const int* tabs[3] = {q, p, n};
    const int* ids = tabs[tree] + b * 128 + node * 32;
    const int c = threadIdx.x * 4;
    float a0 = 0, a1 = 0, a2 = 0, a3 = 0;
    for (int t = 0; t < 32; ++t) {
        if (t + 4 < 32)
            __builtin_prefetch(emb + (size_t)ids[t + 4] * 512 + c, 0, 0);
        const float4 e = *(const float4*)(emb + (size_t)ids[t] * 512 + c);
        a0 += e.x; a1 += e.y; a2 += e.z; a3 += e.w;
    }
    __bf16* o = X + (((size_t)(tree * 512 + b)) * 4 + node) * 512 + c;
    const float s = 1.f / 32.f;
    o[0] = (__bf16)(a0 * s); o[1] = (__bf16)(a1 * s);
    o[2] = (__bf16)(a2 * s); o[3] = (__bf16)(a3 * s);
}

// ---------------------------------------------------------------------------
// bf16 WMMA GEMM: C[M,N] = A[M,K](row stride lda) * WT[N,K]^T (+bias), f32 acc.
// Block = 128 threads = 4 waves, 64x64 block tile; each wave owns a 32x32
// quadrant = 2x2 accumulators, so each A/B fragment feeds 2 WMMAs.
// WT pre-transposed -> both LDS stages and both fragment gathers are pure b128.
// Fragment layout per CDNA5 16-bit A/B spec: lane<16 K{0..7,16..23},
// lane>=16 K{8..15,24..31}.
__global__ __launch_bounds__(128)
void k_gemm_bf16(const __bf16* __restrict__ A, long lda,
                 const __bf16* __restrict__ WT, int N, int K,
                 const float* __restrict__ bias,
                 float* __restrict__ C) {
    __shared__ __bf16 As[64 * LSTR];
    __shared__ __bf16 Bs[64 * LSTR];

    const int tid  = threadIdx.x;
    const int wave = tid >> 5;
    const int lane = tid & 31;
    const int hf   = lane >> 4;
    const int idx  = lane & 15;
    const int wr   = wave >> 1;
    const int wc   = wave & 1;

    const long rowBase = (long)blockIdx.y * 64;
    const int  colBase = blockIdx.x * 64;

    const int srow = tid >> 1;         // 0..63
    const int skc  = (tid & 1) * 16;   // 0 or 16

    const __bf16* aSrc = A + (rowBase + srow) * lda + skc;
    const __bf16* bSrc = WT + (long)(colBase + srow) * K + skc;
#if USE_ASYNC_LDS
    const unsigned aDst = (unsigned)(size_t)&As[srow * LSTR + skc];
    const unsigned bDst = (unsigned)(size_t)&Bs[srow * LSTR + skc];
#endif

    f32x8 acc[2][2] = {};

    for (int k0 = 0; k0 < K; k0 += 32) {
#if USE_ASYNC_LDS
        async_b128(aDst,      aSrc + k0);
        async_b128(aDst + 16, aSrc + k0 + 8);
        async_b128(bDst,      bSrc + k0);
        async_b128(bDst + 16, bSrc + k0 + 8);
        wait_async0();
#else
        *(uint4*)(&As[srow * LSTR + skc])     = *(const uint4*)(aSrc + k0);
        *(uint4*)(&As[srow * LSTR + skc + 8]) = *(const uint4*)(aSrc + k0 + 8);
        *(uint4*)(&Bs[srow * LSTR + skc])     = *(const uint4*)(bSrc + k0);
        *(uint4*)(&Bs[srow * LSTR + skc + 8]) = *(const uint4*)(bSrc + k0 + 8);
#endif
        __syncthreads();

        bf16x16 afr[2], bfr[2];
#pragma unroll
        for (int mi = 0; mi < 2; ++mi) {
            const __bf16* ar = &As[(wr * 32 + mi * 16 + idx) * LSTR + hf * 8];
#pragma unroll
            for (int j = 0; j < 8; ++j) { afr[mi][j] = ar[j]; afr[mi][8 + j] = ar[16 + j]; }
        }
#pragma unroll
        for (int ni = 0; ni < 2; ++ni) {
            const __bf16* br = &Bs[(wc * 32 + ni * 16 + idx) * LSTR + hf * 8];
#pragma unroll
            for (int j = 0; j < 8; ++j) { bfr[ni][j] = br[j]; bfr[ni][8 + j] = br[16 + j]; }
        }
#pragma unroll
        for (int mi = 0; mi < 2; ++mi)
#pragma unroll
            for (int ni = 0; ni < 2; ++ni)
                acc[mi][ni] = __builtin_amdgcn_wmma_f32_16x16x32_bf16(
                    false, afr[mi], false, bfr[ni], (short)0, acc[mi][ni], false, false);
        __syncthreads();
    }

    // C/D layout: VGPR r -> M = r (lanes 0-15) / 8+r (lanes 16-31), N = lane&15
#pragma unroll
    for (int mi = 0; mi < 2; ++mi) {
#pragma unroll
        for (int ni = 0; ni < 2; ++ni) {
            const int col = colBase + wc * 32 + ni * 16 + idx;
            const float bc = bias ? bias[col] : 0.f;
#pragma unroll
            for (int r = 0; r < 8; ++r) {
                const long row = rowBase + wr * 32 + mi * 16 + hf * 8 + r;
                C[row * (long)N + col] = acc[mi][ni][r] + bc;
            }
        }
    }
}

// ---------------------------------------------------------------------------
// Leaf gates: c = sig(i)*tanh(u); h = sig(o)*tanh(c). (forget*0 child = dead)
__global__ __launch_bounds__(128)
void k_leaf_combine(const float* __restrict__ IOU, float* __restrict__ Cl,
                    __bf16* __restrict__ Hl) {
    const int g = blockIdx.x;                 // tree*1536 + b*3 + leaf
    const int tree = g / 1536, rem = g % 1536;
    const int b = rem / 3, leaf = rem % 3;
    const size_t iouRow = ((size_t)(tree * 512 + b) * 4 + leaf) * 1536;
    const int c = threadIdx.x * 4;
#pragma unroll
    for (int j = 0; j < 4; ++j) {
        const float iv = sigf(IOU[iouRow + c + j]);
        const float ov = sigf(IOU[iouRow + 512 + c + j]);
        const float uv = tanhf(IOU[iouRow + 1024 + c + j]);
        const float cc = iv * uv;
        Cl[(size_t)g * 512 + c + j] = cc;
        Hl[(size_t)g * 512 + c + j] = (__bf16)(ov * tanhf(cc));
    }
}

// h_sum over 3 leaves -> bf16
__global__ __launch_bounds__(128)
void k_hsum(const __bf16* __restrict__ Hl, __bf16* __restrict__ Hs) {
    const int r = blockIdx.x;                 // tree*512 + b
    const int c = threadIdx.x * 4;
#pragma unroll
    for (int j = 0; j < 4; ++j) {
        float s = 0.f;
        for (int k = 0; k < 3; ++k)
            s += (float)Hl[((size_t)r * 3 + k) * 512 + c + j];
        Hs[(size_t)r * 512 + c + j] = (__bf16)s;
    }
}

// Root gates: c = i*u + sum_k sig(FH_k + FX) * c_leaf_k  (biases baked into GEMMs)
__global__ __launch_bounds__(128)
void k_root_combine(const float* __restrict__ IOU, const float* __restrict__ IOUH,
                    const float* __restrict__ FX, const float* __restrict__ FH,
                    const float* __restrict__ Cl, float* __restrict__ Cr) {
    const int r = blockIdx.x;                 // tree*512 + b
    const size_t iouRow = ((size_t)r * 4 + 3) * 1536;
    const size_t hRow = (size_t)r * 1536;
    const int c = threadIdx.x * 4;
#pragma unroll
    for (int j = 0; j < 4; ++j) {
        const float iv = sigf(IOU[iouRow + c + j] + IOUH[hRow + c + j]);
        const float uv = tanhf(IOU[iouRow + 1024 + c + j] + IOUH[hRow + 1024 + c + j]);
        float cc = iv * uv;
        const float fx = FX[(size_t)r * 512 + c + j];
        for (int k = 0; k < 3; ++k) {
            const float f = sigf(FH[((size_t)r * 3 + k) * 512 + c + j] + fx);
            cc += f * Cl[((size_t)r * 3 + k) * 512 + c + j];
        }
        Cr[(size_t)r * 512 + c + j] = cc;     // h_root unused downstream
    }
}

// Z[0..B)=qc*pc, Z[B..2B)=qc*nc  (bf16 for sim GEMM)
__global__ __launch_bounds__(128)
void k_zprep(const float* __restrict__ Cr, __bf16* __restrict__ Z) {
    const int p = blockIdx.x / 512, b = blockIdx.x % 512;
    const float* qc = Cr + (size_t)b * 512;
    const float* oc = Cr + (size_t)((p ? 2 : 1) * 512 + b) * 512;
    const int c = threadIdx.x * 4;
#pragma unroll
    for (int j = 0; j < 4; ++j)
        Z[(size_t)blockIdx.x * 512 + c + j] = (__bf16)(qc[c + j] * oc[c + j]);
}

// ---------------------------------------------------------------------------
__device__ __forceinline__ float blockReduce256(float v, float* sb) {
    const int t = threadIdx.x;
    sb[t] = v; __syncthreads();
    for (int s = 128; s > 0; s >>= 1) {
        if (t < s) sb[t] += sb[t + s];
        __syncthreads();
    }
    const float r = sb[0]; __syncthreads();
    return r;
}

// Final: triplet hinge losses + sim hinge. sum_o(sim) = sig(S)·rowsum(Wwp)+const,
// and the constant cancels in (1 - a + b).
__global__ __launch_bounds__(256)
void k_final(const float* __restrict__ POOL, const float* __restrict__ S,
             const float* __restrict__ Wwp, float* __restrict__ out) {
    __shared__ float sb[256];
    const int b = blockIdx.x, t = threadIdx.x;
    const float qa = POOL[(size_t)b * 512 + t];
    const float qb = POOL[(size_t)b * 512 + t + 256];
    float part[6];
#pragma unroll
    for (int k = 0; k < 3; ++k) {
        const float* pa = POOL + ((size_t)(1 + 2 * k) * 512 + b) * 512;
        const float* na = POOL + ((size_t)(2 + 2 * k) * 512 + b) * 512;
        part[2 * k]     = qa * pa[t] + qb * pa[t + 256];
        part[2 * k + 1] = qa * na[t] + qb * na[t + 256];
    }
    float ws = 0.f;
    for (int o = 0; o < 30; ++o) ws += Wwp[t * 30 + o];
    const float s1 = sigf(S[(size_t)b * 256 + t]);
    const float s2 = sigf(S[(size_t)(512 + b) * 256 + t]);
    const float dpart = (s1 - s2) * ws;

    float r[7];
    for (int k = 0; k < 6; ++k) r[k] = blockReduce256(part[k], sb);
    r[6] = blockReduce256(dpart, sb);
    if (t == 0) {
        float loss = 0.f;
        for (int k = 0; k < 3; ++k)
            loss += fmaxf(0.f, 1.f - r[2 * k] + r[2 * k + 1]);
        out[b] = loss + fmaxf(0.f, 1.f - r[6]);
    }
}

// ---------------------------------------------------------------------------
extern "C" void kernel_launch(void* const* d_in, const int* in_sizes, int n_in,
                              void* d_out, int out_size, void* d_ws, size_t ws_size,
                              hipStream_t stream) {
    const int*   q_v   = (const int*)d_in[0];
    const int*   qa0   = (const int*)d_in[1];
    const int*   na0   = (const int*)d_in[2];
    const int*   qa1   = (const int*)d_in[3];
    const int*   na1   = (const int*)d_in[4];
    const int*   qa2   = (const int*)d_in[5];
    const int*   na2   = (const int*)d_in[6];
    const int*   query = (const int*)d_in[7];
    const int*   pos   = (const int*)d_in[8];
    const int*   neg   = (const int*)d_in[9];
    const float* emb   = (const float*)d_in[10];
    const float* Wioux = (const float*)d_in[11];
    const float* bioux = (const float*)d_in[12];
    const float* Wiouh = (const float*)d_in[13];
    const float* biouh = (const float*)d_in[14];
    const float* Wfx   = (const float*)d_in[15];
    const float* bfx   = (const float*)d_in[16];
    const float* Wfh   = (const float*)d_in[17];
    const float* bfh   = (const float*)d_in[18];
    const float* Wwh   = (const float*)d_in[19];
    const float* bwh   = (const float*)d_in[20];
    const float* Wwp   = (const float*)d_in[21];
    float* out = (float*)d_out;

    size_t off = 0;
    char* base = (char*)d_ws;
    auto carve = [&](size_t bytes) -> void* {
        void* r = base + off;
        off = (off + bytes + 255) & ~(size_t)255;
        return r;
    };
    __bf16* WiouxT = (__bf16*)carve(786432 * 2);   // [1536][512]
    __bf16* WiouhT = (__bf16*)carve(786432 * 2);   // [1536][512]
    __bf16* WfxT   = (__bf16*)carve(262144 * 2);   // [512][512]
    __bf16* WfhT   = (__bf16*)carve(262144 * 2);   // [512][512]
    __bf16* WwhT   = (__bf16*)carve(131072 * 2);   // [256][512]
    float*  biasIOU = (float*)carve(1536 * 4);
    float*  POOL    = (float*)carve((size_t)7 * 512 * 512 * 4);
    __bf16* X       = (__bf16*)carve((size_t)6144 * 512 * 2);
    float*  IOUX    = (float*)carve((size_t)6144 * 1536 * 4);
    float*  FX      = (float*)carve((size_t)1536 * 512 * 4);
    float*  Cl      = (float*)carve((size_t)4608 * 512 * 4);
    __bf16* Hl      = (__bf16*)carve((size_t)4608 * 512 * 2);
    __bf16* Hs      = (__bf16*)carve((size_t)1536 * 512 * 2);
    float*  IOUH    = (float*)carve((size_t)1536 * 1536 * 4);
    float*  FH      = (float*)carve((size_t)4608 * 512 * 4);
    float*  Cr      = (float*)carve((size_t)1536 * 512 * 4);
    __bf16* Z       = (__bf16*)carve((size_t)1024 * 512 * 2);
    float*  S       = (float*)carve((size_t)1024 * 256 * 4);
    (void)ws_size; (void)in_sizes; (void)n_in; (void)out_size;

    // weight conversions (fp32 [K][N] -> bf16 [N][K]) + combined iou bias
    k_f2bf_t<<<dim3(1536 / 16, 512 / 16), 256, 0, stream>>>(Wioux, WiouxT, 512, 1536);
    k_f2bf_t<<<dim3(1536 / 16, 512 / 16), 256, 0, stream>>>(Wiouh, WiouhT, 512, 1536);
    k_f2bf_t<<<dim3(512 / 16, 512 / 16), 256, 0, stream>>>(Wfx, WfxT, 512, 512);
    k_f2bf_t<<<dim3(512 / 16, 512 / 16), 256, 0, stream>>>(Wfh, WfhT, 512, 512);
    k_f2bf_t<<<dim3(256 / 16, 512 / 16), 256, 0, stream>>>(Wwh, WwhT, 512, 256);
    k_biasadd<<<(1536 + 255) / 256, 256, 0, stream>>>(bioux, biouh, biasIOU, 1536);

    // embedding gather + mean pools
    k_attr_pool<<<dim3(512, 7), 128, 0, stream>>>(emb, q_v, qa0, na0, qa1, na1, qa2, na2, POOL);
    k_seq_pool<<<dim3(512, 4, 3), 128, 0, stream>>>(emb, query, pos, neg, X);

    // IOU for all nodes (x-part): [6144,512] x [512,1536]
    k_gemm_bf16<<<dim3(1536 / 64, 6144 / 64), 128, 0, stream>>>(X, 512, WiouxT, 1536, 512, biasIOU, IOUX);
    // root fx: strided rows (node 3 of each (tree,b)): [1536,512] x [512,512]
    k_gemm_bf16<<<dim3(512 / 64, 1536 / 64), 128, 0, stream>>>(X + 3 * 512, 2048, WfxT, 512, 512, bfx, FX);

    k_leaf_combine<<<4608, 128, 0, stream>>>(IOUX, Cl, Hl);
    k_hsum<<<1536, 128, 0, stream>>>(Hl, Hs);

    // root h-part iou: [1536,512] x [512,1536]; child forget: [4608,512] x [512,512]
    k_gemm_bf16<<<dim3(1536 / 64, 1536 / 64), 128, 0, stream>>>(Hs, 512, WiouhT, 1536, 512, nullptr, IOUH);
    k_gemm_bf16<<<dim3(512 / 64, 4608 / 64), 128, 0, stream>>>(Hl, 512, WfhT, 512, 512, bfh, FH);

    k_root_combine<<<1536, 128, 0, stream>>>(IOUX, IOUH, FX, FH, Cl, Cr);
    k_zprep<<<1024, 128, 0, stream>>>(Cr, Z);

    // sim hidden: [1024,512] x [512,256]
    k_gemm_bf16<<<dim3(256 / 64, 1024 / 64), 128, 0, stream>>>(Z, 512, WwhT, 256, 512, bwh, S);

    k_final<<<512, 256, 0, stream>>>(POOL, S, Wwp, out);
}
